// MultiHeadAttention_3856880632373
// MI455X (gfx1250) — matrix operand order repaired and checked by reading, not compiled, not run on hardware
//
#include <hip/hip_runtime.h>
#include <hip/hip_bf16.h>
#include <math.h>

// ---------------------------------------------------------------------------
// Types for CDNA5 WMMA (wave32): bf16 A/B fragments, f32 accumulator.
// ---------------------------------------------------------------------------
typedef __attribute__((ext_vector_type(16))) __bf16 v16bf;
typedef __attribute__((ext_vector_type(8)))  __bf16 v8bf;
typedef __attribute__((ext_vector_type(8)))  float  v8f;

union U16 { v16bf v; v8bf h[2]; };
union SF  { v8f v; float f[8]; };

#define WMMA_BF16(a, b, c) \
    __builtin_amdgcn_wmma_f32_16x16x32_bf16(false, (a), false, (b), (short)0, (c), false, false)

// ---------------------------------------------------------------------------
// Problem constants
// ---------------------------------------------------------------------------
#define EMBED 1024
#define NHEAD 16
#define HDIM  64
#define BDIM  2
#define TDIM  2048
#define NEGINF (-1.0e9f)

// ---------------------------------------------------------------------------
// Kernel: f32 -> bf16 convert (row-major copy)
// ---------------------------------------------------------------------------
__global__ __launch_bounds__(256) void convert_bf16(const float* __restrict__ in,
                                                    __bf16* __restrict__ out, size_t n) {
    size_t i = (size_t)blockIdx.x * 256 + threadIdx.x;
    if (i < n) out[i] = (__bf16)in[i];
}

// ---------------------------------------------------------------------------
// Kernel: f32 [rows][cols] -> bf16 transposed [cols][rows]
// ---------------------------------------------------------------------------
__global__ __launch_bounds__(256) void transpose_bf16(const float* __restrict__ in,
                                                      __bf16* __restrict__ out,
                                                      int rows, int cols) {
    size_t i = (size_t)blockIdx.x * 256 + threadIdx.x;
    size_t n = (size_t)rows * cols;
    if (i >= n) return;
    int r = (int)(i / cols);
    int c = (int)(i % cols);
    out[(size_t)c * rows + r] = (__bf16)in[i];
}

// ---------------------------------------------------------------------------
// Kernel: bf16 GEMM   C[M][N] (f32) = A[M][K] (bf16, row-major) @ Bt[N][K]^T
// Bt stores the weight transposed so fragments come from contiguous memory.
//
// Block: 256 threads = 8 waves (4 M x 2 N); block tile 128x128; K-step 32.
// A/B block tiles are staged into double-buffered LDS with async
// global->LDS copies (ASYNCcnt), overlapping the fill of buffer buf^1 with
// WMMA compute from buffer buf. Fragments then come from LDS (DScnt).
// ---------------------------------------------------------------------------
__global__ __launch_bounds__(256) void gemm_bf16(const __bf16* __restrict__ A,
                                                 const __bf16* __restrict__ Bt,
                                                 float* __restrict__ C,
                                                 int M, int N, int K) {
    // 128 rows x 32 K elements, padded to 40 elems (80 B) per row:
    // 16B-chunk aligned, and 20-dword row stride -> 16 consecutive rows land
    // on distinct bank groups (gcd(20,64) cycle = 16) => conflict-free frags.
    constexpr int PADE = 40;
    __shared__ __bf16 Asm[2][128 * PADE];
    __shared__ __bf16 Bsm[2][128 * PADE];

    const int tid  = threadIdx.x;
    const int lane = tid & 31;
    const int w    = tid >> 5;
    const int wm   = w & 3;         // 0..3 -> M offset (32 each)
    const int wn   = w >> 2;        // 0..1 -> N offset (64 each)
    const int l15  = lane & 15;
    const int hi   = lane >> 4;
    const int koffA = hi * 16;      // A-frag per-lane K chunk start (16 contiguous)
    const int koffB = hi * 16;      // B-frag per-lane K chunk start (16 contiguous)

    const int mblock = blockIdx.x * 128;
    const int nblock = blockIdx.y * 128;

    // Async staging: one 128x32 bf16 tile = 512 x 16B chunks; 256 threads
    // issue 2 chunks per tile (A and B) per stage = 4 async ops per thread.
    auto stage = [&](int buf, int kk) {
        const unsigned aBase = (unsigned)(size_t)(&Asm[buf][0]);
        const unsigned bBase = (unsigned)(size_t)(&Bsm[buf][0]);
        #pragma unroll
        for (int q = 0; q < 2; ++q) {
            const int c   = tid + q * 256;          // chunk id in [0,512)
            const int row = c >> 2;
            const int sub = c & 3;                  // 8-element sub-chunk
            const __bf16* ga = A + (size_t)(mblock + row) * K + kk + sub * 8;
            const unsigned la = aBase + (unsigned)(row * PADE + sub * 8) * 2u;
            asm volatile("global_load_async_to_lds_b128 %0, %1, off"
                         :: "v"(la), "v"((unsigned long long)(size_t)ga)
                         : "memory");
            const __bf16* gb = Bt + (size_t)(nblock + row) * K + kk + sub * 8;
            const unsigned lb = bBase + (unsigned)(row * PADE + sub * 8) * 2u;
            asm volatile("global_load_async_to_lds_b128 %0, %1, off"
                         :: "v"(lb), "v"((unsigned long long)(size_t)gb)
                         : "memory");
        }
    };

    v8f acc[2][4];
    for (int i = 0; i < 2; ++i)
        for (int j = 0; j < 4; ++j)
            acc[i][j] = (v8f){0.f, 0.f, 0.f, 0.f, 0.f, 0.f, 0.f, 0.f};

    stage(0, 0);
    asm volatile("s_wait_asynccnt 0x0" ::: "memory");
    __syncthreads();

    int buf = 0;
    for (int kk = 0; kk < K; kk += 32) {
        if (kk + 32 < K) stage(buf ^ 1, kk + 32);   // overlap fill with compute

        v16bf a[2], b[4];
        #pragma unroll
        for (int i = 0; i < 2; ++i) {
            const __bf16* p = &Asm[buf][(wm * 32 + i * 16 + l15) * PADE + koffA];
            U16 u;
            u.h[0] = *(const v8bf*)(p);
            u.h[1] = *(const v8bf*)(p + 8);
            a[i] = u.v;
        }
        #pragma unroll
        for (int j = 0; j < 4; ++j) {
            const __bf16* p = &Bsm[buf][(wn * 64 + j * 16 + l15) * PADE + koffB];
            U16 u;
            u.h[0] = *(const v8bf*)(p);
            u.h[1] = *(const v8bf*)(p + 8);
            b[j] = u.v;
        }
        #pragma unroll
        for (int i = 0; i < 2; ++i)
            #pragma unroll
            for (int j = 0; j < 4; ++j)
                acc[i][j] = WMMA_BF16(a[i], b[j], acc[i][j]);

        asm volatile("s_wait_asynccnt 0x0" ::: "memory");
        __syncthreads();
        buf ^= 1;
    }

    for (int i = 0; i < 2; ++i)
        for (int j = 0; j < 4; ++j) {
            SF sf; sf.v = acc[i][j];
            for (int r = 0; r < 8; ++r) {
                int m = mblock + wm * 32 + i * 16 + r + 8 * hi;
                int n = nblock + wn * 64 + j * 16 + l15;
                C[(size_t)m * N + n] = sf.f[r];
            }
        }
}

// ---------------------------------------------------------------------------
// Kernel: RoPE + head packing.
// qkv f32 [B*T][3*E] -> Q,K bf16 [B][H][T][64] (rotated), Vt bf16 [B][H][64][T]
// ---------------------------------------------------------------------------
__global__ __launch_bounds__(256) void rope_pack(const float* __restrict__ qkv,
                                                 __bf16* __restrict__ Q,
                                                 __bf16* __restrict__ Ko,
                                                 __bf16* __restrict__ Vt) {
    size_t idx = (size_t)blockIdx.x * 256 + threadIdx.x;  // over B*H*T*64
    const int d = (int)(idx & 63);
    size_t r   = idx >> 6;
    const int t = (int)(r % TDIM);
    const int bh = (int)(r / TDIM);
    const int b = bh / NHEAD;
    const int h = bh % NHEAD;

    const float* base = qkv + ((size_t)b * TDIM + t) * (3 * EMBED) + h * HDIM;
    float qv = base[d];
    float kv = base[EMBED + d];
    float vv = base[2 * EMBED + d];

    int i = d & 31;
    // inv_freq = ROPE_BASE^(-(2i)/64)
    float inv_freq = __expf(-(float)(2 * i) / 64.0f * logf(10000.0f));
    float ang = (float)t * inv_freq;
    float c = __cosf(ang);
    float s = __sinf(ang);
    float qrot = (d < 32) ? -base[d + 32] : base[d - 32];
    float krot = (d < 32) ? -base[EMBED + d + 32] : base[EMBED + d - 32];

    size_t o = ((size_t)bh * TDIM + t) * HDIM + d;
    Q[o]  = (__bf16)(qv * c + qrot * s);
    Ko[o] = (__bf16)(kv * c + krot * s);
    Vt[((size_t)bh * HDIM + d) * TDIM + t] = (__bf16)vv;
}

// ---------------------------------------------------------------------------
// Kernel: causal flash attention, one wave per 16 query rows.
// Q,K: [B*H][T][64] bf16 (RoPE'd);  Vt: [B*H][64][T] bf16.
// Output: attn bf16 [B*T][EMBED] (heads re-interleaved) for GEMM2.
// S tile = Q(16x32d) x K^T(32d x 16k) WMMA; online softmax in f32;
// P goes through LDS to swap C-layout -> A-layout; O += P(16x32k) x V(32k x 16d).
// ---------------------------------------------------------------------------
__global__ __launch_bounds__(256) void attn_flash(const __bf16* __restrict__ Q,
                                                  const __bf16* __restrict__ Km,
                                                  const __bf16* __restrict__ Vt,
                                                  __bf16* __restrict__ attn) {
    __shared__ __bf16 Plds[8][16][32];   // per-wave 16x32 bf16 staging (8 KB)

    const int lane = threadIdx.x & 31;
    const int w    = threadIdx.x >> 5;
    const int l15  = lane & 15;
    const int hi   = lane >> 4;
    const int koffA = hi * 16;
    const int koffB = hi * 16;

    const int bh = blockIdx.y;                 // b*NHEAD + h
    const int b  = bh / NHEAD;
    const int h  = bh % NHEAD;
    const int qbase = blockIdx.x * 128 + w * 16;

    const __bf16* Qh = Q  + (size_t)bh * TDIM * HDIM;
    const __bf16* Kh = Km + (size_t)bh * TDIM * HDIM;
    const __bf16* Vh = Vt + (size_t)bh * HDIM * TDIM;

    // Q fragments are invariant over the k loop: 2 frags cover d=0..63.
    v16bf qa[2];
    for (int t = 0; t < 2; ++t) {
        const __bf16* p = Qh + (size_t)(qbase + l15) * HDIM + t * 32 + koffA;
        U16 u;
        u.h[0] = *(const v8bf*)(p);
        u.h[1] = *(const v8bf*)(p + 8);
        qa[t] = u.v;
    }

    float mrow[8], lrow[8];
    SF o[4];
    for (int r = 0; r < 8; ++r) { mrow[r] = -3.0e38f; lrow[r] = 0.f; }
    for (int c = 0; c < 4; ++c)
        for (int r = 0; r < 8; ++r) o[c].f[r] = 0.f;

    const int nkb = (qbase + 47) >> 5;         // causal: k blocks covering k <= qbase+15
    for (int kb = 0; kb < nkb; ++kb) {
        const int k0 = kb * 32;

        // ---- S = Q K^T  (two 16x16 tiles over k, each 2 WMMA over d) ----
        v8f sacc[2];
        sacc[0] = (v8f){0.f,0.f,0.f,0.f,0.f,0.f,0.f,0.f};
        sacc[1] = (v8f){0.f,0.f,0.f,0.f,0.f,0.f,0.f,0.f};
        for (int t = 0; t < 2; ++t) {
            for (int d = 0; d < 2; ++d) {
                const __bf16* p = Kh + (size_t)(k0 + t * 16 + l15) * HDIM + d * 32 + koffB;
                v16bf bk = *(const v16bf*)p;
                sacc[t] = WMMA_BF16(qa[d], bk, sacc[t]);
            }
        }

        // ---- scale + causal mask (C layout: q = r + 8*hi, k = k0 + t*16 + l15) ----
        SF s[2];
        s[0].v = sacc[0]; s[1].v = sacc[1];
        for (int t = 0; t < 2; ++t) {
            int kcol = k0 + t * 16 + l15;
            for (int r = 0; r < 8; ++r) {
                int qrow = qbase + r + 8 * hi;
                float v = s[t].f[r] * 0.125f;          // 1/sqrt(64)
                s[t].f[r] = (kcol > qrow) ? NEGINF : v;
            }
        }

        // ---- online softmax: row stats live per lane-group of 16 ----
        float fscale[8];
        for (int r = 0; r < 8; ++r) {
            float rm = fmaxf(s[0].f[r], s[1].f[r]);
            for (int off = 1; off < 16; off <<= 1)
                rm = fmaxf(rm, __shfl_xor(rm, off, 32));
            float mnew = fmaxf(mrow[r], rm);
            fscale[r] = __expf(mrow[r] - mnew);
            float ps = 0.f;
            for (int t = 0; t < 2; ++t) {
                float p = __expf(s[t].f[r] - mnew);
                s[t].f[r] = p;
                ps += p;
            }
            for (int off = 1; off < 16; off <<= 1)
                ps += __shfl_xor(ps, off, 32);
            lrow[r] = lrow[r] * fscale[r] + ps;
            mrow[r] = mnew;
        }
        for (int c = 0; c < 4; ++c)
            for (int r = 0; r < 8; ++r) o[c].f[r] *= fscale[r];

        // ---- C-layout -> A-layout via per-wave LDS tile ----
        for (int t = 0; t < 2; ++t)
            for (int r = 0; r < 8; ++r)
                Plds[w][r + 8 * hi][t * 16 + l15] = (__bf16)s[t].f[r];
        asm volatile("s_wait_dscnt 0x0" ::: "memory");

        U16 up;
        {
            const __bf16* pp = &Plds[w][l15][koffA];
            up.h[0] = *(const v8bf*)(pp);
            up.h[1] = *(const v8bf*)(pp + 8);
        }
        v16bf pa = up.v;

        // ---- O += P @ V  (V^T rows are contiguous in k) ----
        for (int c = 0; c < 4; ++c) {
            const __bf16* vp = Vh + (size_t)(c * 16 + l15) * TDIM + k0 + koffB;
            v16bf bv = *(const v16bf*)vp;
            v8f oc;
            SF tmp = o[c];
            oc = tmp.v;
            oc = WMMA_BF16(pa, bv, oc);
            o[c].v = oc;
        }
        asm volatile("s_wait_dscnt 0x0" ::: "memory");  // before next iter overwrites Plds
    }

    // ---- normalize and scatter heads back to [B*T][EMBED] ----
    for (int c = 0; c < 4; ++c) {
        for (int r = 0; r < 8; ++r) {
            int q = qbase + r + 8 * hi;
            float val = o[c].f[r] / lrow[r];
            attn[((size_t)b * TDIM + q) * EMBED + h * HDIM + c * 16 + l15] = (__bf16)val;
        }
    }
}

// ---------------------------------------------------------------------------
// Host launcher
// ---------------------------------------------------------------------------
extern "C" void kernel_launch(void* const* d_in, const int* in_sizes, int n_in,
                              void* d_out, int out_size, void* d_ws, size_t ws_size,
                              hipStream_t stream) {
    const float* x     = (const float*)d_in[0];   // [B*T][1024] f32
    const float* w_qkv = (const float*)d_in[1];   // [1024][3072] f32
    const float* w_out = (const float*)d_in[2];   // [1024][1024] f32
    float* out = (float*)d_out;                    // [B*T][1024] f32

    const size_t BT = (size_t)BDIM * TDIM;         // 4096

    // Workspace carve-up (256B aligned)
    char* ws = (char*)d_ws;
    size_t off = 0;
    auto carve = [&](size_t bytes) -> char* {
        char* p = ws + off;
        off += (bytes + 255) & ~(size_t)255;
        return p;
    };
    __bf16* xb     = (__bf16*)carve(BT * EMBED * sizeof(__bf16));                // 8 MB
    __bf16* wqkvT  = (__bf16*)carve((size_t)3 * EMBED * EMBED * sizeof(__bf16)); // 6 MB
    __bf16* woutT  = (__bf16*)carve((size_t)EMBED * EMBED * sizeof(__bf16));     // 2 MB
    float*  qkv    = (float*) carve(BT * 3 * EMBED * sizeof(float));             // 48 MB
    __bf16* Qr     = (__bf16*)carve(BT * EMBED * sizeof(__bf16));                // 8 MB
    __bf16* Kr     = (__bf16*)carve(BT * EMBED * sizeof(__bf16));                // 8 MB
    __bf16* Vt     = (__bf16*)carve(BT * EMBED * sizeof(__bf16));                // 8 MB
    __bf16* attn   = (__bf16*)carve(BT * EMBED * sizeof(__bf16));                // 8 MB
    (void)ws_size;

    // 1) Precision conversion / weight transposition
    {
        size_t n = BT * EMBED;
        convert_bf16<<<dim3((unsigned)((n + 255) / 256)), dim3(256), 0, stream>>>(x, xb, n);
    }
    {
        size_t n = (size_t)EMBED * 3 * EMBED;
        transpose_bf16<<<dim3((unsigned)((n + 255) / 256)), dim3(256), 0, stream>>>(
            w_qkv, wqkvT, EMBED, 3 * EMBED);
    }
    {
        size_t n = (size_t)EMBED * EMBED;
        transpose_bf16<<<dim3((unsigned)((n + 255) / 256)), dim3(256), 0, stream>>>(
            w_out, woutT, EMBED, EMBED);
    }

    // 2) QKV projection: [4096][3072] = xb @ w_qkv
    gemm_bf16<<<dim3(BT / 128, (3 * EMBED) / 128), dim3(256), 0, stream>>>(
        xb, wqkvT, qkv, (int)BT, 3 * EMBED, EMBED);

    // 3) RoPE + head packing (Q,K rotated; V transposed per head)
    {
        size_t n = (size_t)BDIM * NHEAD * TDIM * HDIM;
        rope_pack<<<dim3((unsigned)(n / 256)), dim3(256), 0, stream>>>(qkv, Qr, Kr, Vt);
    }

    // 4) Causal flash attention -> attn bf16 [4096][1024]
    attn_flash<<<dim3(TDIM / 128, BDIM * NHEAD), dim3(256), 0, stream>>>(Qr, Kr, Vt, attn);

    // 5) Output projection: out = attn @ w_out
    gemm_bf16<<<dim3(BT / 128, EMBED / 128), dim3(256), 0, stream>>>(
        attn, woutT, out, (int)BT, EMBED, EMBED);
}
